// GAT_77730318123060
// MI455X (gfx1250) — matrix operand order repaired and checked by reading, not compiled, not run on hardware
//
#include <hip/hip_runtime.h>
#include <hip/hip_bf16.h>

typedef unsigned short u16;
typedef unsigned int u32;
typedef __bf16 bf16_t;
typedef bf16_t v16bf __attribute__((ext_vector_type(16)));
typedef float v8f __attribute__((ext_vector_type(8)));
typedef unsigned int u32x4 __attribute__((ext_vector_type(4)));
typedef int i32x4 __attribute__((ext_vector_type(4)));
typedef int i32x8 __attribute__((ext_vector_type(8)));

#define NEG_SLOPE 0.2f
#define TPB 256

// ---------- helpers ----------
__device__ __forceinline__ u16 f2bf(float f) {
    u32 u = __float_as_uint(f);
    u32 r = u + 0x7FFFu + ((u >> 16) & 1u);   // round-to-nearest-even
    return (u16)(r >> 16);
}
// order-preserving float <-> uint encoding for atomicMax-based segment max
__device__ __forceinline__ u32 encf(float f) {
    u32 u = __float_as_uint(f);
    return (u & 0x80000000u) ? ~u : (u | 0x80000000u);
}
__device__ __forceinline__ float decf(u32 e) {
    u32 u = (e & 0x80000000u) ? (e & 0x7FFFFFFFu) : ~e;
    return __uint_as_float(u);
}

// ---------- fill ----------
__global__ void fill_u32(u32* __restrict__ p, u32 v, int n) {
    int i = blockIdx.x * TPB + threadIdx.x;
    if (i < n) p[i] = v;
}

// ---------- pack A (fp32 [rows, K] -> bf16 WMMA A-fragment layout, [mt][ks][512]) ----------
__global__ void packA(const float* __restrict__ X, u16* __restrict__ Ap,
                      int Mtiles, int KS, int ld, int nRows) {
    int i = blockIdx.x * TPB + threadIdx.x;
    int total = Mtiles * KS * 512;
    if (i >= total) return;
    int e    = i & 15;
    int lane = (i >> 4) & 31;
    int frag = i >> 9;
    int ks   = frag % KS;
    int mt   = frag / KS;
    int v    = e >> 1;
    int k = ks * 32 + ((v >> 2) << 4) + ((lane >> 4) << 3) + ((v & 3) << 1) + (e & 1);
    int m = mt * 16 + (lane & 15);
    float f = (m < nRows) ? X[(size_t)m * ld + k] : 0.0f;
    Ap[i] = f2bf(f);
}

// same as packA but applies relu(x + bias[k]) (layer-2 input)
__global__ void packA_relu_bias(const float* __restrict__ X, const float* __restrict__ bias,
                                u16* __restrict__ Ap, int Mtiles, int KS, int ld, int nRows) {
    int i = blockIdx.x * TPB + threadIdx.x;
    int total = Mtiles * KS * 512;
    if (i >= total) return;
    int e    = i & 15;
    int lane = (i >> 4) & 31;
    int frag = i >> 9;
    int ks   = frag % KS;
    int mt   = frag / KS;
    int v    = e >> 1;
    int k = ks * 32 + ((v >> 2) << 4) + ((lane >> 4) << 3) + ((v & 3) << 1) + (e & 1);
    int m = mt * 16 + (lane & 15);
    float f = 0.0f;
    if (m < nRows) f = fmaxf(X[(size_t)m * ld + k] + bias[k], 0.0f);
    Ap[i] = f2bf(f);
}

// ---------- pack B (W [Nout, K] fp32 -> bf16 WMMA B-fragment layout, B = W^T) ----------
// layout [ks][nt][512] so a wave's adjacent n-tiles are contiguous at fixed ks.
// B 32x16: lane n (0-15) holds K=0..7,16..23, lane n+16 holds K=8..15,24..31 of column n
__global__ void packB(const float* __restrict__ W, u16* __restrict__ Bp,
                      int Ntiles, int KS, int ldw) {
    int i = blockIdx.x * TPB + threadIdx.x;
    int total = Ntiles * KS * 512;
    if (i >= total) return;
    int e    = i & 15;
    int lane = (i >> 4) & 31;
    int frag = i >> 9;
    int nt   = frag % Ntiles;
    int ks   = frag / Ntiles;
    int k = ks * 32 + ((e >> 3) << 4) + ((lane >> 4) << 3) + (e & 7);
    int n = nt * 16 + (lane & 15);
    Bp[i] = f2bf(W[(size_t)n * ldw + k]);
}

// ---------- layer-1 WMMA GEMM, TDM-staged A, 2 M-tiles x 4 N-tiles per wave ----------
// One block (8 waves) covers M-tiles {2b, 2b+1} x all 32 N-tiles; wave w owns n-tiles
// [4w, 4w+4). 32 KB of A ([mt][ks][512] contiguous) loaded once via tensor_load_to_lds;
// per K-step each wave does 8 WMMAs (2 A-fragments x 4 B-fragments) -> 4 B128 loads feed
// 2 WMMAs each, doubling matrix-op density per load-wait vs 1 M-tile.
__global__ void gemm1_wmma_tdm(const u16* __restrict__ Ap, const u16* __restrict__ Bp,
                               float* __restrict__ C, int ldc, int nRows, int Mtiles) {
    __shared__ __align__(128) u16 Asm[2 * 16 * 512];    // 32 KB
    const int KS = 16, NT = 32;
    int mt0  = blockIdx.x * 2;
    int wv   = threadIdx.x >> 5;
    int lane = threadIdx.x & 31;

    if (wv == 0) {
        // ---- Tensor DMA Descriptor (D#): 1-D tile of 16384 x 2B elements ----
        unsigned long long ga = (unsigned long long)(Ap + (size_t)mt0 * KS * 512);
        u32 lds = (u32)(size_t)&Asm[0];
        u32x4 g0;
        g0.x = 1u;                                            // count=1 valid descriptor
        g0.y = lds;                                           // lds_addr [63:32]
        g0.z = (u32)ga;                                       // global_addr [95:64]
        g0.w = (u32)((ga >> 32) & 0x01FFFFFFu) | (2u << 30);  // addr[120:96], type=2
        i32x8 g1;
        g1[0] = (int)(1u << 16);         // wg_mask=0, data_size=1 (2 bytes)
        g1[1] = (int)(16384u << 16);     // tensor_dim0[15:0] at [63:48]
        g1[2] = (int)(1u << 16);         // tensor_dim0 hi=0, tensor_dim1=1
        g1[3] = (int)(16384u << 16);     // tile_dim0=16384 at [127:112]
        g1[4] = 0;                       // tile_dim1=0 (unused), tile_dim2=0
        g1[5] = 16384;                   // tensor_dim0_stride[31:0]
        g1[6] = 0;
        g1[7] = 0;
        i32x4 zz4 = {0, 0, 0, 0};
        i32x8 zz8 = {0, 0, 0, 0, 0, 0, 0, 0};
        __builtin_amdgcn_tensor_load_to_lds(g0, g1, zz4, zz4, zz8, 0);
        __builtin_amdgcn_s_wait_tensorcnt(0);
    }
    __syncthreads();

    int nt0 = wv * 4;
    v8f c00 = {}, c01 = {}, c02 = {}, c03 = {};
    v8f c10 = {}, c11 = {}, c12 = {}, c13 = {};
#pragma unroll 2
    for (int ks = 0; ks < KS; ++ks) {
        v16bf a0 = *reinterpret_cast<const v16bf*>(&Asm[ks * 512 + lane * 16]);
        v16bf a1 = *reinterpret_cast<const v16bf*>(&Asm[8192 + ks * 512 + lane * 16]);
        const u16* Bb = Bp + ((size_t)ks * NT + nt0) * 512 + lane * 16;
        v16bf b0 = *reinterpret_cast<const v16bf*>(Bb);
        v16bf b1 = *reinterpret_cast<const v16bf*>(Bb + 512);
        v16bf b2 = *reinterpret_cast<const v16bf*>(Bb + 1024);
        v16bf b3 = *reinterpret_cast<const v16bf*>(Bb + 1536);
        c00 = __builtin_amdgcn_wmma_f32_16x16x32_bf16(false, a0, false, b0, (short)0, c00, false, false);
        c10 = __builtin_amdgcn_wmma_f32_16x16x32_bf16(false, a1, false, b0, (short)0, c10, false, false);
        c01 = __builtin_amdgcn_wmma_f32_16x16x32_bf16(false, a0, false, b1, (short)0, c01, false, false);
        c11 = __builtin_amdgcn_wmma_f32_16x16x32_bf16(false, a1, false, b1, (short)0, c11, false, false);
        c02 = __builtin_amdgcn_wmma_f32_16x16x32_bf16(false, a0, false, b2, (short)0, c02, false, false);
        c12 = __builtin_amdgcn_wmma_f32_16x16x32_bf16(false, a1, false, b2, (short)0, c12, false, false);
        c03 = __builtin_amdgcn_wmma_f32_16x16x32_bf16(false, a0, false, b3, (short)0, c03, false, false);
        c13 = __builtin_amdgcn_wmma_f32_16x16x32_bf16(false, a1, false, b3, (short)0, c13, false, false);
    }
    // C layout: lanes 0-15 N=lane, M=v; lanes 16-31 N=lane-16, M=8+v
    int ncol = (lane & 15);
    int moff = ((lane >> 4) << 3);
#pragma unroll
    for (int mi = 0; mi < 2; ++mi) {
        int mt = mt0 + mi;
        if (mt >= Mtiles) break;
        v8f* cc0 = (mi == 0) ? &c00 : &c10;
        v8f* cc1 = (mi == 0) ? &c01 : &c11;
        v8f* cc2 = (mi == 0) ? &c02 : &c12;
        v8f* cc3 = (mi == 0) ? &c03 : &c13;
        int mbase = mt * 16 + moff;
#pragma unroll
        for (int v = 0; v < 8; ++v) {
            int row = mbase + v;
            if (row < nRows) {
                float* cr = C + (size_t)row * ldc + ncol;
                cr[(nt0 + 0) * 16] = (*cc0)[v];
                cr[(nt0 + 1) * 16] = (*cc1)[v];
                cr[(nt0 + 2) * 16] = (*cc2)[v];
                cr[(nt0 + 3) * 16] = (*cc3)[v];
            }
        }
    }
}

// ---------- generic WMMA bf16 GEMM (direct from global), one wave per 16x16 tile ----------
__global__ void gemm_wmma_bf16(const u16* __restrict__ Ap, const u16* __restrict__ Bp,
                               float* __restrict__ C, int Mtiles, int Ntiles, int KS,
                               int ldc, int nRows) {
    int wave = (blockIdx.x * TPB + (int)threadIdx.x) >> 5;
    int lane = threadIdx.x & 31;
    if (wave >= Mtiles * Ntiles) return;      // wave-uniform: EXEC stays all-ones for WMMA
    int mt = wave / Ntiles;
    int nt = wave % Ntiles;
    const u16* Ab = Ap + (size_t)mt * KS * 512;
    v8f c = {};
#pragma unroll 4
    for (int ks = 0; ks < KS; ++ks) {
        v16bf a = *reinterpret_cast<const v16bf*>(Ab + (size_t)ks * 512 + lane * 16);
        v16bf b = *reinterpret_cast<const v16bf*>(Bp + ((size_t)ks * Ntiles + nt) * 512 + lane * 16);
        c = __builtin_amdgcn_wmma_f32_16x16x32_bf16(false, a, false, b, (short)0, c,
                                                    false, false);
    }
    int n     = nt * 16 + (lane & 15);
    int mbase = mt * 16 + ((lane >> 4) << 3);
#pragma unroll
    for (int v = 0; v < 8; ++v) {
        int row = mbase + v;
        if (row < nRows) C[(size_t)row * ldc + n] = c[v];
    }
}

// ---------- per-node attention logits: a_s[n,h] = h[n,h,:].att_s[h,:], same for a_d ----------
__global__ void logits(const float* __restrict__ Hm, const float* __restrict__ att_s,
                       const float* __restrict__ att_d, float* __restrict__ as_,
                       float* __restrict__ ad_, int n_nodes, int heads, int ch) {
    int i = blockIdx.x * TPB + threadIdx.x;
    if (i >= n_nodes * heads) return;
    int node = i / heads, hd = i % heads;
    const float* hp = Hm + (size_t)node * heads * ch + (size_t)hd * ch;
    const float* sp = att_s + (size_t)hd * ch;
    const float* dp = att_d + (size_t)hd * ch;
    float ss = 0.f, dd = 0.f;
    for (int c = 0; c < ch; ++c) { float v = hp[c]; ss += v * sp[c]; dd += v * dp[c]; }
    as_[i] = ss;
    ad_[i] = dd;
}

// ---------- edge pass: segment max of leaky-relu logits (uint-encoded atomicMax) ----------
__global__ void edge_max(const long long* __restrict__ ei, int E, int EE,
                         const float* __restrict__ as_, const float* __restrict__ ad_,
                         u32* __restrict__ m, int H) {
    int i = blockIdx.x * TPB + threadIdx.x;
    if (i >= EE * H) return;
    int edge = i / H, h = i % H;
    int src, dst;
    if (edge < E) { src = (int)ei[edge]; dst = (int)ei[E + edge]; }
    else          { src = dst = edge - E; }
    float e = as_[src * H + h] + ad_[dst * H + h];
    e = (e >= 0.f) ? e : NEG_SLOPE * e;
    atomicMax(m + dst * H + h, encf(e));
}

// ---------- edge pass: segment sum of exp(e - max) ----------
__global__ void edge_sum(const long long* __restrict__ ei, int E, int EE,
                         const float* __restrict__ as_, const float* __restrict__ ad_,
                         const u32* __restrict__ m, float* __restrict__ z, int H) {
    int i = blockIdx.x * TPB + threadIdx.x;
    if (i >= EE * H) return;
    int edge = i / H, h = i % H;
    int src, dst;
    if (edge < E) { src = (int)ei[edge]; dst = (int)ei[E + edge]; }
    else          { src = dst = edge - E; }
    float e = as_[src * H + h] + ad_[dst * H + h];
    e = (e >= 0.f) ? e : NEG_SLOPE * e;
    atomicAdd(z + dst * H + h, __expf(e - decf(m[dst * H + h])));
}

// ---------- layer-1 aggregation: out[dst, h*64+c] += alpha[h] * h1[src, h*64+c] ----------
// one block per edge; alphas (8 heads) computed once into LDS; 2 channels/thread
__global__ void agg1(const long long* __restrict__ ei, int E, int EE,
                     const float* __restrict__ as_, const float* __restrict__ ad_,
                     const u32* __restrict__ m, const float* __restrict__ z,
                     const float* __restrict__ H1b, float* __restrict__ out) {
    __shared__ float alpha_s[8];
    int edge = blockIdx.x;
    int t = threadIdx.x;
    int src, dst;
    if (edge < E) { src = (int)ei[edge]; dst = (int)ei[E + edge]; }
    else          { src = dst = edge - E; }
    if (t < 8) {
        float e = as_[src * 8 + t] + ad_[dst * 8 + t];
        e = (e >= 0.f) ? e : NEG_SLOPE * e;
        float p = __expf(e - decf(m[dst * 8 + t]));
        alpha_s[t] = p / z[dst * 8 + t];
    }
    __syncthreads();
    int c0 = t * 2;                                   // 512 channels / 256 threads
    float a = alpha_s[c0 >> 6];                       // head = channel / 64
    float2 hv = *reinterpret_cast<const float2*>(H1b + (size_t)src * 512 + c0);
    atomicAdd(out + (size_t)dst * 512 + c0,     a * hv.x);
    atomicAdd(out + (size_t)dst * 512 + c0 + 1, a * hv.y);
}

// ---------- layer-2 aggregation (H=1, C=16): thread per (edge, channel) ----------
__global__ void agg2(const long long* __restrict__ ei, int E, int EE,
                     const float* __restrict__ as_, const float* __restrict__ ad_,
                     const u32* __restrict__ m, const float* __restrict__ z,
                     const float* __restrict__ H2b, float* __restrict__ acc) {
    int i = blockIdx.x * TPB + threadIdx.x;
    if (i >= EE * 16) return;
    int edge = i >> 4, c = i & 15;
    int src, dst;
    if (edge < E) { src = (int)ei[edge]; dst = (int)ei[E + edge]; }
    else          { src = dst = edge - E; }
    float e = as_[src] + ad_[dst];
    e = (e >= 0.f) ? e : NEG_SLOPE * e;
    float alpha = __expf(e - decf(m[dst])) / z[dst];
    atomicAdd(acc + (size_t)dst * 16 + c, alpha * H2b[(size_t)src * 16 + c]);
}

// ---------- final: out = acc + b2 ----------
__global__ void final_bias(const float* __restrict__ acc, const float* __restrict__ b2,
                           float* __restrict__ out, int n) {
    int i = blockIdx.x * TPB + threadIdx.x;
    if (i < n) out[i] = acc[i] + b2[i & 15];
}

static inline int cdiv(long long a, long long b) { return (int)((a + b - 1) / b); }

extern "C" void kernel_launch(void* const* d_in, const int* in_sizes, int n_in,
                              void* d_out, int out_size, void* d_ws, size_t ws_size,
                              hipStream_t stream) {
    const float*      x      = (const float*)d_in[0];
    const long long*  ei     = (const long long*)d_in[1];   // int64 [2, E]
    const float*      W1     = (const float*)d_in[2];       // [512, 512]
    const float*      att_s1 = (const float*)d_in[3];       // [8, 64]
    const float*      att_d1 = (const float*)d_in[4];
    const float*      b1     = (const float*)d_in[5];       // [512]
    const float*      W2     = (const float*)d_in[6];       // [16, 512]
    const float*      att_s2 = (const float*)d_in[7];       // [16]
    const float*      att_d2 = (const float*)d_in[8];
    const float*      b2     = (const float*)d_in[9];       // [16]

    const int IN  = 512;                 // layer-1 K
    const int C1  = 512;                 // layer-1 out channels (8 heads * 64)
    const int C2  = 16;                  // layer-2 out channels
    const int KS  = IN / 32;             // 16 K-steps of 32
    const int N   = in_sizes[0] / IN;    // 20000
    const int E   = in_sizes[1] / 2;     // 320000
    const int EE  = E + N;               // + self loops
    const int Mt  = (N + 15) / 16;       // 1250
    const int Nt1 = C1 / 16;             // 32
    const int Nt2 = C2 / 16;             // 1

    // ---- workspace layout (256B aligned) ----
    char* w = (char*)d_ws;
    size_t off = 0;
    auto alloc = [&](size_t bytes) { size_t o = off; off = (off + bytes + 255) & ~(size_t)255; return o; };
    u16*   A1p  = (u16*)  (w + alloc((size_t)(Mt + 1) * KS * 512 * 2));   // +1 tile pad for 2-tile TDM
    u16*   B1p  = (u16*)  (w + alloc((size_t)Nt1 * KS * 512 * 2));
    float* h1   = (float*)(w + alloc((size_t)N * C1 * 4));
    float* out1 = (float*)(w + alloc((size_t)N * C1 * 4));
    float* as1  = (float*)(w + alloc((size_t)N * 8 * 4));
    float* ad1  = (float*)(w + alloc((size_t)N * 8 * 4));
    u32*   m1   = (u32*)  (w + alloc((size_t)N * 8 * 4));
    float* z1   = (float*)(w + alloc((size_t)N * 8 * 4));
    u16*   A2p  = (u16*)  (w + alloc((size_t)Mt * KS * 512 * 2));
    u16*   B2p  = (u16*)  (w + alloc((size_t)Nt2 * KS * 512 * 2));
    float* h2   = (float*)(w + alloc((size_t)N * C2 * 4));
    float* as2  = (float*)(w + alloc((size_t)N * 4));
    float* ad2  = (float*)(w + alloc((size_t)N * 4));
    u32*   m2   = (u32*)  (w + alloc((size_t)N * 4));
    float* z2   = (float*)(w + alloc((size_t)N * 4));
    float* acc2 = (float*)(w + alloc((size_t)N * C2 * 4));
    (void)ws_size; (void)n_in;

    // ---- zero accumulators (every call: deterministic across graph replays) ----
    fill_u32<<<cdiv((size_t)N * C1, TPB), TPB, 0, stream>>>((u32*)out1, 0u, N * C1);
    fill_u32<<<cdiv(N * 8, TPB), TPB, 0, stream>>>(m1, 0u, N * 8);
    fill_u32<<<cdiv(N * 8, TPB), TPB, 0, stream>>>((u32*)z1, 0u, N * 8);
    fill_u32<<<cdiv(N, TPB), TPB, 0, stream>>>(m2, 0u, N);
    fill_u32<<<cdiv(N, TPB), TPB, 0, stream>>>((u32*)z2, 0u, N);
    fill_u32<<<cdiv(N * C2, TPB), TPB, 0, stream>>>((u32*)acc2, 0u, N * C2);

    // ---- layer 1: pack + WMMA GEMM (TDM-staged A, 2 M-tiles/block)  h1 = x @ W1^T ----
    int a1_elems = Mt * KS * 512;
    packA<<<cdiv(a1_elems, TPB), TPB, 0, stream>>>(x, A1p, Mt, KS, IN, N);
    packB<<<cdiv(Nt1 * KS * 512, TPB), TPB, 0, stream>>>(W1, B1p, Nt1, KS, IN);
    gemm1_wmma_tdm<<<cdiv(Mt, 2), TPB, 0, stream>>>(A1p, B1p, h1, C1, N, Mt);

    // ---- layer 1: attention ----
    logits<<<cdiv(N * 8, TPB), TPB, 0, stream>>>(h1, att_s1, att_d1, as1, ad1, N, 8, 64);
    edge_max<<<cdiv(EE * 8, TPB), TPB, 0, stream>>>(ei, E, EE, as1, ad1, m1, 8);
    edge_sum<<<cdiv(EE * 8, TPB), TPB, 0, stream>>>(ei, E, EE, as1, ad1, m1, z1, 8);
    agg1<<<EE, TPB, 0, stream>>>(ei, E, EE, as1, ad1, m1, z1, h1, out1);

    // ---- layer 2: relu(out1 + b1) packed to bf16, WMMA GEMM  h2 = hrelu @ W2^T ----
    packA_relu_bias<<<cdiv(a1_elems, TPB), TPB, 0, stream>>>(out1, b1, A2p, Mt, KS, C1, N);
    packB<<<cdiv(Nt2 * KS * 512, TPB), TPB, 0, stream>>>(W2, B2p, Nt2, KS, C1);
    gemm_wmma_bf16<<<cdiv((size_t)Mt * Nt2 * 32, TPB), TPB, 0, stream>>>(
        A2p, B2p, h2, Mt, Nt2, KS, C2, N);

    // ---- layer 2: attention ----
    logits<<<cdiv(N, TPB), TPB, 0, stream>>>(h2, att_s2, att_d2, as2, ad2, N, 1, 16);
    edge_max<<<cdiv(EE, TPB), TPB, 0, stream>>>(ei, E, EE, as2, ad2, m2, 1);
    edge_sum<<<cdiv(EE, TPB), TPB, 0, stream>>>(ei, E, EE, as2, ad2, m2, z2, 1);
    agg2<<<cdiv(EE * 16, TPB), TPB, 0, stream>>>(ei, E, EE, as2, ad2, m2, z2, h2, acc2);

    // ---- output ----
    final_bias<<<cdiv(N * C2, TPB), TPB, 0, stream>>>(acc2, b2, (float*)d_out, N * C2);
}
